// HyperSimCLRLoss_73108933312729
// MI455X (gfx1250) — matrix operand order repaired and checked by reading, not compiled, not run on hardware
//
#include <hip/hip_runtime.h>

typedef __attribute__((ext_vector_type(16))) __bf16 v16bf;
typedef __attribute__((ext_vector_type(8)))  __bf16 v8bf;
typedef __attribute__((ext_vector_type(8)))  float  v8f;

union V16U { v16bf v; v8bf h[2]; };

#define KP      544        // 512 (z) + 8 (x0) padded to multiple of 32
#define KFLAT   512
#define HH      8
#define DD      64
#define CSPLIT  8
#define RSTRIDE (2 * KP)   // interleaved row: [hi 544 | lo 544] u16 elems (2176 B)
#define TILEOFF (16 * RSTRIDE) // 16 rows ahead, in elems (34816 B)
#define L2E     1.4426950408889634f
#define LN2     0.6931471805599453f
#define NEG_BIG (-1e30f)

typedef unsigned short u16;

__device__ __forceinline__ u16 f2bf_rne(float f) {
    unsigned int u = __float_as_uint(f);
    unsigned int r = u + 0x7FFFu + ((u >> 16) & 1u);   // round-to-nearest-even
    return (u16)(r >> 16);
}
__device__ __forceinline__ float bf2f(u16 b) {
    return __uint_as_float(((unsigned int)b) << 16);
}

// ---------------------------------------------------------------------------
// Kernel 1: x0[n,h] = sqrt(sum_d z[n,h,d]^2 + beta[h])
// ---------------------------------------------------------------------------
__global__ void x0_kernel(const float* __restrict__ zi, const float* __restrict__ zj,
                          const float* __restrict__ beta, float* __restrict__ x0,
                          int N, int Bn) {
    int t = blockIdx.x * blockDim.x + threadIdx.x;
    if (t >= N * HH) return;
    int n = t >> 3, h = t & 7;
    const float* p = (n < Bn ? zi + (size_t)n * KFLAT
                             : zj + (size_t)(n - Bn) * KFLAT) + h * DD;
    const float4* p4 = (const float4*)p;
    float s = 0.f;
#pragma unroll
    for (int i = 0; i < DD / 4; ++i) {
        float4 q = p4[i];
        s += q.x * q.x + q.y * q.y + q.z * q.z + q.w * q.w;
    }
    x0[t] = sqrtf(s + beta[h]);
}

// ---------------------------------------------------------------------------
// Kernel 2: build augmented bf16 hi/lo matrices, hi/lo interleaved per row:
//   row n of Amat: [hi(544) | lo(544)],  A[n] = 4*[z_flat, +x0, 0-pad]
//   row n of Bmat: [hi(544) | lo(544)],  B[n] =   [z_flat, -x0, 0-pad]
//   A . B^T = 4*(space - time) = sim (constant term cancels in the loss)
// ---------------------------------------------------------------------------
__global__ void convert_kernel(const float* __restrict__ zi, const float* __restrict__ zj,
                               const float* __restrict__ x0,
                               u16* __restrict__ Amat, u16* __restrict__ Bmat,
                               int N, int Bn) {
    int n = blockIdx.x;
    const float* zrow = (n < Bn) ? zi + (size_t)n * KFLAT
                                 : zj + (size_t)(n - Bn) * KFLAT;
    for (int k = threadIdx.x; k < KP; k += blockDim.x) {
        float v;
        if (k < KFLAT)            v = zrow[k];
        else if (k < KFLAT + HH)  v = x0[n * HH + (k - KFLAT)];
        else                      v = 0.f;
        float av = 4.f * v;                                   // 2/temp, exact pow2
        float bv = (k >= KFLAT && k < KFLAT + HH) ? -v : v;   // time-part sign

        u16 ah = f2bf_rne(av);
        u16 al = f2bf_rne(av - bf2f(ah));
        u16 bh = f2bf_rne(bv);
        u16 bl = f2bf_rne(bv - bf2f(bh));

        size_t idx = (size_t)n * RSTRIDE + k;
        Amat[idx] = ah; Amat[idx + KP] = al;
        Bmat[idx] = bh; Bmat[idx + KP] = bl;
    }
}

// ---------------------------------------------------------------------------
// Kernel 3: flash GEMM + online softmax stats.
// One wave owns a 32-row block x (N/CSPLIT)-column chunk, processed as
// 32x64 output blocks: 8 accumulators, 17 k-steps x 24 bf16 WMMAs per step
// (bf16x3 split: hi*hi + hi*lo + lo*hi, fp32 accumulate).
// Single base pointer per matrix; all tile/hi-lo offsets are immediates.
// ---------------------------------------------------------------------------
__global__ __launch_bounds__(32)
void flash_kernel(const u16* __restrict__ Amat, const u16* __restrict__ Bmat,
                  float* __restrict__ pm, float* __restrict__ ps, int N) {
    const int lane = threadIdx.x & 31;
    const int half = lane >> 4;     // 0: lanes 0-15, 1: lanes 16-31
    const int lcol = lane & 15;

    const int strips = N / 32;
    const int strip  = blockIdx.x % strips;
    const int chunk  = blockIdx.x / strips;
    const int rbase  = strip * 32;
    const int cwidth = N / CSPLIT;
    const int cbase  = chunk * cwidth;

    // A 16-bit fragment layout: lane holds row M = lane%16;
    // elems 0-7 = K[kb+half*8..], elems 8-15 = K[kb+16+half*8..].
    const u16* ap0 = Amat + (size_t)(rbase + lcol) * RSTRIDE + half * 8;
    // B 16-bit fragment layout: lane holds col N = lane%16;
    // elems 0-7 = K[kb+half*8..], elems 8-15 = K[kb+16+half*8..].
    const u16* bp0 = Bmat + (size_t)(cbase + lcol) * RSTRIDE + half * 8;

    float m[2][8], ssum[2][8];
#pragma unroll
    for (int s = 0; s < 2; ++s)
#pragma unroll
        for (int r = 0; r < 8; ++r) { m[s][r] = NEG_BIG; ssum[s][r] = 0.f; }

    for (int cb = 0; cb < cwidth; cb += 64) {
        v8f acc[2][4] = {};

        const u16* a = ap0;
        const u16* b = bp0 + (size_t)cb * RSTRIDE;

#pragma unroll 1   // keep accumulators pinned; no pipeliner copy churn
        for (int ks = 0; ks < KP / 32; ++ks) {
            V16U ah[2], al[2];
#pragma unroll
            for (int s = 0; s < 2; ++s) {
                ah[s].h[0] = *(const v8bf*)(a + s * TILEOFF);
                ah[s].h[1] = *(const v8bf*)(a + s * TILEOFF + 16);
                al[s].h[0] = *(const v8bf*)(a + s * TILEOFF + KP);
                al[s].h[1] = *(const v8bf*)(a + s * TILEOFF + KP + 16);
            }
#pragma unroll
            for (int t = 0; t < 4; ++t) {
                V16U ubh, ubl;
                ubh.h[0] = *(const v8bf*)(b + t * TILEOFF);
                ubh.h[1] = *(const v8bf*)(b + t * TILEOFF + 16);
                ubl.h[0] = *(const v8bf*)(b + t * TILEOFF + KP);
                ubl.h[1] = *(const v8bf*)(b + t * TILEOFF + KP + 16);
                v16bf bh = ubh.v;
                v16bf bl = ubl.v;
#pragma unroll
                for (int s = 0; s < 2; ++s) {
                    acc[s][t] = __builtin_amdgcn_wmma_f32_16x16x32_bf16(
                                    false, ah[s].v, false, bh, (short)0, acc[s][t], false, false);
                    acc[s][t] = __builtin_amdgcn_wmma_f32_16x16x32_bf16(
                                    false, ah[s].v, false, bl, (short)0, acc[s][t], false, false);
                    acc[s][t] = __builtin_amdgcn_wmma_f32_16x16x32_bf16(
                                    false, al[s].v, false, bh, (short)0, acc[s][t], false, false);
                }
            }
            a += 32; b += 32;
        }

        // Online softmax update: C/D layout = VGPR r holds row (r + half*8), N = lane%16.
#pragma unroll
        for (int s = 0; s < 2; ++s) {
#pragma unroll
            for (int r = 0; r < 8; ++r) {
                const int row = rbase + s * 16 + r + half * 8;
                float v0 = acc[s][0][r], v1 = acc[s][1][r],
                      v2 = acc[s][2][r], v3 = acc[s][3][r];
                const int c0 = cbase + cb + lcol;
                if (c0      == row) v0 = NEG_BIG;   // mask diagonal
                if (c0 + 16 == row) v1 = NEG_BIG;
                if (c0 + 32 == row) v2 = NEG_BIG;
                if (c0 + 48 == row) v3 = NEG_BIG;

                float bm = fmaxf(fmaxf(v0, v1), fmaxf(v2, v3));
                bm = fmaxf(bm, __shfl_xor(bm, 1));
                bm = fmaxf(bm, __shfl_xor(bm, 2));
                bm = fmaxf(bm, __shfl_xor(bm, 4));
                bm = fmaxf(bm, __shfl_xor(bm, 8));

                float mn = fmaxf(m[s][r], bm);
                float e = __builtin_amdgcn_exp2f((v0 - mn) * L2E)
                        + __builtin_amdgcn_exp2f((v1 - mn) * L2E)
                        + __builtin_amdgcn_exp2f((v2 - mn) * L2E)
                        + __builtin_amdgcn_exp2f((v3 - mn) * L2E);
                e += __shfl_xor(e, 1);
                e += __shfl_xor(e, 2);
                e += __shfl_xor(e, 4);
                e += __shfl_xor(e, 8);

                ssum[s][r] = ssum[s][r] * __builtin_amdgcn_exp2f((m[s][r] - mn) * L2E) + e;
                m[s][r] = mn;
            }
        }
    }

    if (lcol == 0) {   // lanes 0 and 16 each publish their half's rows
#pragma unroll
        for (int s = 0; s < 2; ++s)
#pragma unroll
            for (int r = 0; r < 8; ++r) {
                const int row = rbase + s * 16 + r + half * 8;
                pm[(size_t)chunk * N + row] = m[s][r];
                ps[(size_t)chunk * N + row] = ssum[s][r];
            }
    }
}

// ---------------------------------------------------------------------------
// Kernel 4: merge chunk partials -> lse; exact fp32 positive logit; row loss.
// ---------------------------------------------------------------------------
__global__ void combine_kernel(const float* __restrict__ zi, const float* __restrict__ zj,
                               const float* __restrict__ x0,
                               const float* __restrict__ pm, const float* __restrict__ ps,
                               float* __restrict__ rowloss, int N, int Bn) {
    int n = blockIdx.x * blockDim.x + threadIdx.x;
    if (n >= N) return;
    int p = n + Bn; if (p >= N) p -= N;

    const float* a = (n < Bn) ? zi + (size_t)n * KFLAT : zj + (size_t)(n - Bn) * KFLAT;
    const float* b = (p < Bn) ? zi + (size_t)p * KFLAT : zj + (size_t)(p - Bn) * KFLAT;
    const float4* a4 = (const float4*)a;
    const float4* b4 = (const float4*)b;

    float dot = 0.f;
#pragma unroll 4
    for (int k = 0; k < KFLAT / 4; ++k) {
        float4 qa = a4[k], qb = b4[k];
        dot += qa.x * qb.x + qa.y * qb.y + qa.z * qb.z + qa.w * qb.w;
    }
    float tdot = 0.f;
#pragma unroll
    for (int h = 0; h < HH; ++h) tdot += x0[n * HH + h] * x0[p * HH + h];

    float pos = 4.f * (dot - tdot);   // exact fp32 positive logit (constant cancels)

    float mm = pm[n];
#pragma unroll
    for (int c = 1; c < CSPLIT; ++c) mm = fmaxf(mm, pm[(size_t)c * N + n]);
    float ss = 0.f;
#pragma unroll
    for (int c = 0; c < CSPLIT; ++c)
        ss += ps[(size_t)c * N + n] * __builtin_amdgcn_exp2f((pm[(size_t)c * N + n] - mm) * L2E);

    float lse = mm + __builtin_amdgcn_logf(ss) * LN2;   // v_log = log2 -> scale to ln
    rowloss[n] = lse - pos;
}

// ---------------------------------------------------------------------------
// Kernel 5: deterministic single-block reduction -> scalar loss.
// ---------------------------------------------------------------------------
__global__ void reduce_kernel(const float* __restrict__ rowloss, float* __restrict__ out, int N) {
    __shared__ float sm[256];
    float acc = 0.f;
    for (int i = threadIdx.x; i < N; i += 256) acc += rowloss[i];
    sm[threadIdx.x] = acc;
    __syncthreads();
    for (int st = 128; st > 0; st >>= 1) {
        if ((int)threadIdx.x < st) sm[threadIdx.x] += sm[threadIdx.x + st];
        __syncthreads();
    }
    if (threadIdx.x == 0) out[0] = sm[0] / (float)N;
}

// ---------------------------------------------------------------------------
static inline size_t align256(size_t x) { return (x + 255) & ~(size_t)255; }

extern "C" void kernel_launch(void* const* d_in, const int* in_sizes, int n_in,
                              void* d_out, int out_size, void* d_ws, size_t ws_size,
                              hipStream_t stream) {
    const float* zi   = (const float*)d_in[0];
    const float* zj   = (const float*)d_in[1];
    const float* beta = (const float*)d_in[2];

    const int Bn = in_sizes[0] / KFLAT;   // 4096
    const int N  = 2 * Bn;                // 8192

    // Workspace carve-up (~36 MB total)
    char* w = (char*)d_ws;
    const size_t matBytes = align256((size_t)N * RSTRIDE * sizeof(u16));
    u16* Amat = (u16*)w;  w += matBytes;
    u16* Bmat = (u16*)w;  w += matBytes;
    float* x0 = (float*)w;       w += align256((size_t)N * HH * sizeof(float));
    float* pm = (float*)w;       w += align256((size_t)CSPLIT * N * sizeof(float));
    float* ps = (float*)w;       w += align256((size_t)CSPLIT * N * sizeof(float));
    float* rowloss = (float*)w;  w += align256((size_t)N * sizeof(float));
    (void)ws_size; (void)n_in; (void)out_size;

    // 1) time coordinates
    x0_kernel<<<(N * HH + 255) / 256, 256, 0, stream>>>(zi, zj, beta, x0, N, Bn);
    // 2) augmented bf16 hi/lo matrices (hi/lo interleaved rows)
    convert_kernel<<<N, 256, 0, stream>>>(zi, zj, x0, Amat, Bmat, N, Bn);
    // 3) flash GEMM + online softmax partials (one wave per 32-row strip x chunk)
    flash_kernel<<<(N / 32) * CSPLIT, 32, 0, stream>>>(Amat, Bmat, pm, ps, N);
    // 4) merge partials, exact positive logit, per-row loss
    combine_kernel<<<(N + 255) / 256, 256, 0, stream>>>(zi, zj, x0, pm, ps, rowloss, N, Bn);
    // 5) deterministic scalar reduction
    reduce_kernel<<<1, 256, 0, stream>>>(rowloss, (float*)d_out, N);
}
